// TimeCondAttention_84344567759269
// MI455X (gfx1250) — compile-verified, matrix operand order checked by reading, not run on hardware
//
#include <hip/hip_runtime.h>

typedef _Float16 h8_t  __attribute__((ext_vector_type(8)));
typedef _Float16 v16h  __attribute__((ext_vector_type(16)));
typedef float    v8f   __attribute__((ext_vector_type(8)));

#define BATCH 2
#define SEQ   2048
#define DIMSZ 512
#define HEADS 8
#define DHEAD 64
#define NEGV  1.0e6f

// ---------------------------------------------------------------------------
// helpers
// ---------------------------------------------------------------------------
__device__ __forceinline__ v16h mk16(h8_t lo, h8_t hi) {
    v16h r;
#pragma unroll
    for (int i = 0; i < 8; ++i) { r[i] = lo[i]; r[i + 8] = hi[i]; }
    return r;
}

// ---------------------------------------------------------------------------
// f32 -> f16 convert
// ---------------------------------------------------------------------------
__global__ void cvt_f16_kernel(const float* __restrict__ in,
                               _Float16* __restrict__ out, int n) {
    int i = blockIdx.x * 256 + threadIdx.x;
    if (i < n) out[i] = (_Float16)in[i];
}

// ---------------------------------------------------------------------------
// FiLM: film[b, o] = b_time[o] + sum_k silu(time[b,k]) * w_time[o,k]
// ---------------------------------------------------------------------------
__global__ void film_kernel(const float* __restrict__ timep,
                            const float* __restrict__ w_time,
                            const float* __restrict__ b_time,
                            float* __restrict__ film) {
    int b = blockIdx.x;
    __shared__ float st[DIMSZ];
    for (int i = threadIdx.x; i < DIMSZ; i += 256) {
        float t = timep[b * DIMSZ + i];
        st[i] = t / (1.0f + __expf(-t));   // silu
    }
    __syncthreads();
    for (int o = threadIdx.x; o < 2 * DIMSZ; o += 256) {
        const float* wr = w_time + (size_t)o * DIMSZ;
        float acc = b_time[o];
        for (int k = 0; k < DIMSZ; ++k) acc = fmaf(st[k], wr[k], acc);
        film[b * 2 * DIMSZ + o] = acc;
    }
}

// ---------------------------------------------------------------------------
// LayerNorm + FiLM + seq-mask + fold dim_head^-0.5, emit f16; also x -> f16
// one block (256 thr) per token, 2 elements per thread
// ---------------------------------------------------------------------------
__global__ void ln_film_kernel(const float* __restrict__ x,
                               const float* __restrict__ gamma,
                               const float* __restrict__ film,
                               const float* __restrict__ seq_mask,
                               _Float16* __restrict__ xn16,
                               _Float16* __restrict__ x16) {
    int tok = blockIdx.x;                 // 0 .. B*SEQ-1
    int b   = tok / SEQ;
    int tid = threadIdx.x;
    const float* xr = x + (size_t)tok * DIMSZ;
    float v0 = xr[tid], v1 = xr[tid + 256];
    float s  = v0 + v1;
    float sq = v0 * v0 + v1 * v1;
#pragma unroll
    for (int off = 16; off; off >>= 1) {
        s  += __shfl_xor(s,  off, 32);
        sq += __shfl_xor(sq, off, 32);
    }
    __shared__ float ssum[8], ssq[8];
    int wv = tid >> 5, ln = tid & 31;
    if (ln == 0) { ssum[wv] = s; ssq[wv] = sq; }
    __syncthreads();
    float tot = 0.f, totq = 0.f;
#pragma unroll
    for (int i = 0; i < 8; ++i) { tot += ssum[i]; totq += ssq[i]; }
    float mu   = tot * (1.0f / DIMSZ);
    float var  = totq * (1.0f / DIMSZ) - mu * mu;
    float rinv = rsqrtf(var + 1e-5f);
    float mk   = seq_mask[tok];
    const float* fb = film + (size_t)b * 2 * DIMSZ;
#pragma unroll
    for (int e = 0; e < 2; ++e) {
        int   d  = tid + e * 256;
        float v  = (e == 0) ? v0 : v1;
        float xn = (v - mu) * rinv * gamma[d];
        xn = xn * (fb[d] + 1.0f) + fb[DIMSZ + d];
        xn = xn * mk * 0.125f;            // fold 64^-0.5 (only q uses xn)
        xn16[(size_t)tok * DIMSZ + d] = (_Float16)xn;
        x16 [(size_t)tok * DIMSZ + d] = (_Float16)v;
    }
}

// ---------------------------------------------------------------------------
// WMMA GEMM, 16x64 tile per wave (A fragment reused across 4 N-tiles),
// f16 output with arbitrary (batched) output strides:
//   Out[b*obs + (row%npb)*ors + col*ocs] = (A[row,:] . W[col,:])
// ---------------------------------------------------------------------------
__global__ void gemm_f16out_kernel(const _Float16* __restrict__ A,
                                   const _Float16* __restrict__ W,
                                   _Float16* __restrict__ Out,
                                   int M, int NOUT, int K,
                                   int npb, long long obs, int ors, int ocs) {
    int wave = blockIdx.x * (blockDim.x >> 5) + (threadIdx.x >> 5);
    int tiles_n = NOUT >> 6;                   // 64 columns per wave
    int tm = wave / tiles_n, tn = wave % tiles_n;
    if (tm >= (M >> 4)) return;
    int lane = threadIdx.x & 31;
    int l15 = lane & 15, halfsel = lane >> 4, base0 = halfsel ? 8 : 0;
    int rowbase = tm << 4, colbase = tn << 6;
    const _Float16* arow = A + (size_t)(rowbase + l15) * K;
    const _Float16* wrow[4];
#pragma unroll
    for (int c = 0; c < 4; ++c)
        wrow[c] = W + (size_t)(colbase + c * 16 + l15) * K;
    v8f acc[4];
#pragma unroll
    for (int c = 0; c < 4; ++c) acc[c] = (v8f){};
    for (int kk = 0; kk < K; kk += 32) {
        v16h a = mk16(*(const h8_t*)(arow + kk + base0),
                      *(const h8_t*)(arow + kk + 16 + base0));
#pragma unroll
        for (int c = 0; c < 4; ++c) {
            v16h bf = mk16(*(const h8_t*)(wrow[c] + kk + base0),
                           *(const h8_t*)(wrow[c] + kk + 16 + base0));
            acc[c] = __builtin_amdgcn_wmma_f32_16x16x32_f16(
                false, a, false, bf, (short)0, acc[c], false, false);
        }
    }
#pragma unroll
    for (int c = 0; c < 4; ++c) {
#pragma unroll
        for (int r = 0; r < 8; ++r) {
            int grow = rowbase + r + halfsel * 8;
            int bidx = grow / npb, rloc = grow % npb;
            size_t addr = (size_t)bidx * obs + (size_t)rloc * ors
                        + (size_t)(colbase + c * 16 + l15) * ocs;
            Out[addr] = (_Float16)acc[c][r];
        }
    }
}

// WMMA GEMM (16x64 tile), f32 output [M, NOUT], multiplied by per-row mask
__global__ void gemm_f32mask_kernel(const _Float16* __restrict__ A,
                                    const _Float16* __restrict__ W,
                                    const float* __restrict__ seq_mask,
                                    float* __restrict__ Out,
                                    int M, int NOUT, int K) {
    int wave = blockIdx.x * (blockDim.x >> 5) + (threadIdx.x >> 5);
    int tiles_n = NOUT >> 6;
    int tm = wave / tiles_n, tn = wave % tiles_n;
    if (tm >= (M >> 4)) return;
    int lane = threadIdx.x & 31;
    int l15 = lane & 15, halfsel = lane >> 4, base0 = halfsel ? 8 : 0;
    int rowbase = tm << 4, colbase = tn << 6;
    const _Float16* arow = A + (size_t)(rowbase + l15) * K;
    const _Float16* wrow[4];
#pragma unroll
    for (int c = 0; c < 4; ++c)
        wrow[c] = W + (size_t)(colbase + c * 16 + l15) * K;
    v8f acc[4];
#pragma unroll
    for (int c = 0; c < 4; ++c) acc[c] = (v8f){};
    for (int kk = 0; kk < K; kk += 32) {
        v16h a = mk16(*(const h8_t*)(arow + kk + base0),
                      *(const h8_t*)(arow + kk + 16 + base0));
#pragma unroll
        for (int c = 0; c < 4; ++c) {
            v16h bf = mk16(*(const h8_t*)(wrow[c] + kk + base0),
                           *(const h8_t*)(wrow[c] + kk + 16 + base0));
            acc[c] = __builtin_amdgcn_wmma_f32_16x16x32_f16(
                false, a, false, bf, (short)0, acc[c], false, false);
        }
    }
#pragma unroll
    for (int r = 0; r < 8; ++r) {
        int grow = rowbase + r + halfsel * 8;
        float mk = seq_mask[grow];
#pragma unroll
        for (int c = 0; c < 4; ++c)
            Out[(size_t)grow * NOUT + colbase + c * 16 + l15] = acc[c][r] * mk;
    }
}

// ---------------------------------------------------------------------------
// Flash attention: one wave per 16-row query tile; 32 keys per step.
//   q16, k16: [B, SEQ, H*64] f16      vT: [B, H*64, SEQ] f16
//   bias: [B, H, SEQ, SEQ] f32        ao: [B, SEQ, H*64] f16
// LDS P-slab is private per wave; DS ops are in-order within a wave, so no
// block barrier is needed -- explicit s_wait_dscnt 0 fences the transpose.
// ---------------------------------------------------------------------------
__global__ void attn_kernel(const _Float16* __restrict__ q16,
                            const _Float16* __restrict__ k16,
                            const _Float16* __restrict__ vT,
                            const float* __restrict__ bias,
                            const float* __restrict__ seq_mask,
                            _Float16* __restrict__ ao) {
    __shared__ __align__(16) _Float16 plds[4][16 * 32];
    int wib   = threadIdx.x >> 5;
    int lane  = threadIdx.x & 31;
    int gwave = blockIdx.x * 4 + wib;
    int qtile = gwave & 127;              // SEQ/16 = 128
    int h     = (gwave >> 7) & 7;
    int b     = gwave >> 10;
    int qbase = qtile << 4;
    int l15 = lane & 15, halfsel = lane >> 4, base0 = halfsel ? 8 : 0;
    size_t bN   = (size_t)b * SEQ;
    int    hoff = h * DHEAD;
    _Float16* pl = plds[wib];

    // Q fragments (d 0..31, d 32..63), scale already folded into xn
    const _Float16* qr = q16 + (bN + qbase + l15) * DIMSZ + hoff;
    v16h aq0 = mk16(*(const h8_t*)(qr + base0),      *(const h8_t*)(qr + 16 + base0));
    v16h aq1 = mk16(*(const h8_t*)(qr + 32 + base0), *(const h8_t*)(qr + 48 + base0));

    float mi[8];
#pragma unroll
    for (int r = 0; r < 8; ++r) mi[r] = seq_mask[bN + qbase + r + halfsel * 8];

    float mrow[8], lrow[8];
    v8f   Oacc[4];
#pragma unroll
    for (int r = 0; r < 8; ++r) { mrow[r] = -1.0e30f; lrow[r] = 0.0f; }
#pragma unroll
    for (int t = 0; t < 4; ++t) Oacc[t] = (v8f){};

    const float* bbase = bias + ((size_t)(b * HEADS + h) * SEQ + qbase) * SEQ;

    for (int jb = 0; jb < SEQ / 32; ++jb) {
        int kb = jb * 32;
        // ---- S = Q . K^T  (two 16-key tiles) ----
        const _Float16* kr0 = k16 + (bN + kb + l15) * DIMSZ + hoff;
        const _Float16* kr1 = kr0 + 16 * DIMSZ;
        v16h b00 = mk16(*(const h8_t*)(kr0 + base0),      *(const h8_t*)(kr0 + 16 + base0));
        v16h b01 = mk16(*(const h8_t*)(kr0 + 32 + base0), *(const h8_t*)(kr0 + 48 + base0));
        v16h b10 = mk16(*(const h8_t*)(kr1 + base0),      *(const h8_t*)(kr1 + 16 + base0));
        v16h b11 = mk16(*(const h8_t*)(kr1 + 32 + base0), *(const h8_t*)(kr1 + 48 + base0));
        v8f z = {};
        v8f S0 = __builtin_amdgcn_wmma_f32_16x16x32_f16(false, aq0, false, b00, (short)0, z,  false, false);
        S0     = __builtin_amdgcn_wmma_f32_16x16x32_f16(false, aq1, false, b01, (short)0, S0, false, false);
        v8f S1 = __builtin_amdgcn_wmma_f32_16x16x32_f16(false, aq0, false, b10, (short)0, z,  false, false);
        S1     = __builtin_amdgcn_wmma_f32_16x16x32_f16(false, aq1, false, b11, (short)0, S1, false, false);

        float mj0 = seq_mask[bN + kb + l15];
        float mj1 = seq_mask[bN + kb + 16 + l15];

        // ---- bias add + pair mask + online softmax (row = lanes across N) ----
#pragma unroll
        for (int r = 0; r < 8; ++r) {
            const float* bm = bbase + (size_t)(r + halfsel * 8) * SEQ + kb;
            if (jb + 1 < SEQ / 32 && r == 0) __builtin_prefetch(bm + 32, 0, 0);
            float s0 = S0[r] + bm[l15]      - (1.0f - mi[r] * mj0) * NEGV;
            float s1 = S1[r] + bm[16 + l15] - (1.0f - mi[r] * mj1) * NEGV;
            float tmax = fmaxf(s0, s1);
#pragma unroll
            for (int off = 1; off < 16; off <<= 1)
                tmax = fmaxf(tmax, __shfl_xor(tmax, off, 16));
            float mnew  = fmaxf(mrow[r], tmax);
            float alpha = __expf(mrow[r] - mnew);
            float p0 = __expf(s0 - mnew);
            float p1 = __expf(s1 - mnew);
            float rs = p0 + p1;
#pragma unroll
            for (int off = 1; off < 16; off <<= 1)
                rs += __shfl_xor(rs, off, 16);
            lrow[r] = lrow[r] * alpha + rs;
            mrow[r] = mnew;
#pragma unroll
            for (int t = 0; t < 4; ++t) Oacc[t][r] *= alpha;
            int m = r + halfsel * 8;
            pl[m * 32 + l15]      = (_Float16)p0;   // C-layout -> LDS
            pl[m * 32 + 16 + l15] = (_Float16)p1;
        }
        // DS ops are in-order within a wave; fence store->load explicitly.
        asm volatile("s_wait_dscnt 0" ::: "memory");

        // ---- P A-fragment from LDS (A-layout) ----
        const _Float16* pr = pl + l15 * 32;
        v16h pa = mk16(*(const h8_t*)(pr + base0), *(const h8_t*)(pr + 16 + base0));

        // ---- O += P . V  (V stored transposed: keys contiguous) ----
#pragma unroll
        for (int t = 0; t < 4; ++t) {
            const _Float16* vr = vT + ((size_t)b * DIMSZ + hoff + t * 16 + l15) * SEQ + kb;
            v16h vb = mk16(*(const h8_t*)(vr + base0), *(const h8_t*)(vr + 16 + base0));
            Oacc[t] = __builtin_amdgcn_wmma_f32_16x16x32_f16(
                false, pa, false, vb, (short)0, Oacc[t], false, false);
        }
    }

    // ---- normalize + store ----
#pragma unroll
    for (int r = 0; r < 8; ++r) {
        float inv = (lrow[r] > 0.0f) ? (1.0f / lrow[r]) : 0.0f;
        size_t row = bN + qbase + r + halfsel * 8;
#pragma unroll
        for (int t = 0; t < 4; ++t)
            ao[row * DIMSZ + hoff + t * 16 + l15] = (_Float16)(Oacc[t][r] * inv);
    }
}

// ---------------------------------------------------------------------------
// launcher
// ---------------------------------------------------------------------------
extern "C" void kernel_launch(void* const* d_in, const int* in_sizes, int n_in,
                              void* d_out, int out_size, void* d_ws, size_t ws_size,
                              hipStream_t stream) {
    const float* x     = (const float*)d_in[0];
    const float* timep = (const float*)d_in[1];
    const float* bias  = (const float*)d_in[2];
    const float* mask  = (const float*)d_in[3];
    const float* gamma = (const float*)d_in[4];
    const float* wtime = (const float*)d_in[5];
    const float* btime = (const float*)d_in[6];
    const float* wq    = (const float*)d_in[7];
    const float* wkv   = (const float*)d_in[8];
    const float* wout  = (const float*)d_in[9];
    float* out = (float*)d_out;

    uint8_t* ws = (uint8_t*)d_ws;
    size_t off = 0;
    auto alloc = [&](size_t bytes) -> uint8_t* {
        uint8_t* p = ws + off;
        off += (bytes + 255) & ~(size_t)255;
        return p;
    };
    const size_t ACT = (size_t)BATCH * SEQ * DIMSZ;        // 2M elems
    float*    film  = (float*)   alloc(BATCH * 2 * DIMSZ * 4);
    _Float16* xn16  = (_Float16*)alloc(ACT * 2);
    _Float16* x16   = (_Float16*)alloc(ACT * 2);
    _Float16* wq16  = (_Float16*)alloc((size_t)DIMSZ * DIMSZ * 2);
    _Float16* wkv16 = (_Float16*)alloc((size_t)2 * DIMSZ * DIMSZ * 2);
    _Float16* wo16  = (_Float16*)alloc((size_t)DIMSZ * DIMSZ * 2);
    _Float16* q16   = (_Float16*)alloc(ACT * 2);
    _Float16* k16   = (_Float16*)alloc(ACT * 2);
    _Float16* vT16  = (_Float16*)alloc(ACT * 2);
    _Float16* ao16  = (_Float16*)alloc(ACT * 2);

    // weight converts
    cvt_f16_kernel<<<(DIMSZ * DIMSZ + 255) / 256, 256, 0, stream>>>(wq, wq16, DIMSZ * DIMSZ);
    cvt_f16_kernel<<<(2 * DIMSZ * DIMSZ + 255) / 256, 256, 0, stream>>>(wkv, wkv16, 2 * DIMSZ * DIMSZ);
    cvt_f16_kernel<<<(DIMSZ * DIMSZ + 255) / 256, 256, 0, stream>>>(wout, wo16, DIMSZ * DIMSZ);

    // FiLM params, then fused LN+FiLM+mask
    film_kernel<<<BATCH, 256, 0, stream>>>(timep, wtime, btime, film);
    ln_film_kernel<<<BATCH * SEQ, 256, 0, stream>>>(x, gamma, film, mask, xn16, x16);

    const int M = BATCH * SEQ;                    // 4096
    const int tiles64 = (M / 16) * (DIMSZ / 64);  // 2048 waves -> 512 blocks
    // q = xn @ wq^T        -> [B,SEQ,512]
    gemm_f16out_kernel<<<tiles64 / 4, 128, 0, stream>>>(
        xn16, wq16, q16, M, DIMSZ, DIMSZ, SEQ, (long long)SEQ * DIMSZ, DIMSZ, 1);
    // k = x @ wkv[0:512]^T -> [B,SEQ,512]
    gemm_f16out_kernel<<<tiles64 / 4, 128, 0, stream>>>(
        x16, wkv16, k16, M, DIMSZ, DIMSZ, SEQ, (long long)SEQ * DIMSZ, DIMSZ, 1);
    // v = x @ wkv[512:]^T  -> stored TRANSPOSED [B, 512, SEQ]
    gemm_f16out_kernel<<<tiles64 / 4, 128, 0, stream>>>(
        x16, wkv16 + (size_t)DIMSZ * DIMSZ, vT16, M, DIMSZ, DIMSZ,
        SEQ, (long long)DIMSZ * SEQ, 1, SEQ);

    // flash attention: B*H*(SEQ/16) = 2048 waves, 4 waves / block
    attn_kernel<<<BATCH * HEADS * (SEQ / 16) / 4, 128, 0, stream>>>(
        q16, k16, vT16, bias, mask, ao16);

    // out = (ao @ wout^T) * mask  -> f32
    gemm_f32mask_kernel<<<tiles64 / 4, 128, 0, stream>>>(
        ao16, wo16, mask, out, M, DIMSZ, DIMSZ);
}
